// LSTMEncoderLayer_43696997270109
// MI455X (gfx1250) — matrix operand order repaired and checked by reading, not compiled, not run on hardware
//
#include <hip/hip_runtime.h>

// LSTM encoder: B=64, T=512, D=512, U=1024.
// Per-timestep fused GEMM  g = [x_t | h_{t-1}] * [Wx; Wh] + b  with
// v_wmma_f32_16x16x32_f16. Weights are staged into LDS by the Tensor Data
// Mover (tensor_load_to_lds, TENSORcnt), double-buffered across K-chunks so
// the DMA overlaps WMMA compute and B is loaded once per block (not per wave).

typedef __attribute__((ext_vector_type(16))) _Float16 v16h;
typedef __attribute__((ext_vector_type(8)))  _Float16 v8h;
typedef __attribute__((ext_vector_type(8)))  float    v8f;
typedef __attribute__((ext_vector_type(4)))  unsigned int v4u;
typedef __attribute__((ext_vector_type(4)))  int      v4i;
typedef __attribute__((ext_vector_type(8)))  int      v8i_t;

#define LSTM_B   64
#define LSTM_T   512
#define LSTM_D   512
#define LSTM_U   1024
#define LSTM_K   (LSTM_D + LSTM_U)   // 1536
#define LSTM_4U  (4 * LSTM_U)        // 4096

#define KCHUNK   256                 // K elements staged per chunk
#define NCHUNK   (LSTM_K / KCHUNK)   // 6
#define KT_PER_CHUNK (KCHUNK / 32)   // 8 k-tiles per chunk
// LDS chunk buffer: 64 rows (4 gates x 16 u) x 256 f16 = 32 KB; x2 buffers
#define CH_HALF  16384               // f16 elements per buffer

// ---------------- prep kernels ----------------

__global__ void k_cvt_x(const float* __restrict__ x, _Float16* __restrict__ x16, int n) {
    int i = blockIdx.x * 256 + threadIdx.x;
    if (i < n) x16[i] = (_Float16)x[i];
}

// Pack [Wx; Wh] (K=1536 x N=4096, row-major over K) into column-major-per-n
// f16 layout Wp[n*1536 + k] so B fragments / TDM rows are contiguous.
__global__ void k_pack_w(const float* __restrict__ Wx, const float* __restrict__ Wh,
                         _Float16* __restrict__ Wp) {
    int n = blockIdx.x;  // 0..4095
    for (int k = threadIdx.x; k < LSTM_K; k += blockDim.x) {
        float v = (k < LSTM_D) ? Wx[(size_t)k * LSTM_4U + n]
                               : Wh[(size_t)(k - LSTM_D) * LSTM_4U + n];
        Wp[(size_t)n * LSTM_K + k] = (_Float16)v;
    }
}

__global__ void k_init(_Float16* __restrict__ h16, float* __restrict__ c32) {
    int i = blockIdx.x * 256 + threadIdx.x;  // B*U = 65536 total
    h16[i] = (_Float16)0.0f;
    c32[i] = 0.0f;
}

// ---------------- fragment loaders ----------------

static __device__ __forceinline__ v16h load_a(const _Float16* __restrict__ xrow,
                                              const _Float16* __restrict__ hrow,
                                              int kt, int lh) {
    const _Float16* ap = ((kt < LSTM_D) ? (xrow + kt) : (hrow + (kt - LSTM_D))) + lh * 8;
    union { v16h v; v8h h[2]; } A;
    A.h[0] = *(const v8h*)(ap);
    A.h[1] = *(const v8h*)(ap + 16);
    return A.v;
}

static __device__ __forceinline__ v16h load_b_lds(const _Float16* bp) {
    union { v16h v; v8h h[2]; } Bv;
    Bv.h[0] = *(const v8h*)(bp);
    Bv.h[1] = *(const v8h*)(bp + 16);
    return Bv.v;
}

#define WMMA_F16(A, Bm, C) \
    __builtin_amdgcn_wmma_f32_16x16x32_f16(false, (A), false, (Bm), (short)0, (C), false, false)

// ---------------- TDM: 2D tile (16 rows x 512B) global -> LDS ----------------
// D# per CDNA5 ISA. data_size=8B: tile 64x16 elements,
// row stride 384 elements (= 1536 f16 = one Wp row).
// This toolchain's builtin takes 6 args: (v4u, v8i, v4i, v4i, v8i, i32 cpol).

static __device__ __forceinline__ void tdm_load_gate(unsigned lds_off, const _Float16* gptr) {
    unsigned long long ga = (unsigned long long)(uintptr_t)gptr;
    v4u g0 = { 1u,                                              // count=1, user desc
               lds_off,                                         // lds_addr (bytes)
               (unsigned)(ga & 0xffffffffu),                    // global_addr[31:0]
               (unsigned)((ga >> 32) & 0x01ffffffu) | 0x80000000u };  // [56:32] | type=2
    v8i_t g1 = { (int)(3u << 16),                 // wg_mask=0, data_size=3 (8B), no flags
                 (int)((64u & 0xffffu) << 16),    // tensor_dim0 = 64 (low 16)
                 (int)((16u & 0xffffu) << 16),    // tensor_dim0 hi=0 | tensor_dim1 = 16
                 (int)(64u << 16),                // tensor_dim1 hi=0 | tile_dim0 = 64
                 (int)16,                         // tile_dim1 = 16 | tile_dim2 = 0
                 (int)384,                        // tensor_dim0_stride[31:0] = 384
                 0,                               // stride0 hi | stride1 lo
                 0 };                             // stride1 hi
    v4i g2 = { 0, 0, 0, 0 };
    v4i g3 = { 0, 0, 0, 0 };
    v8i_t g4 = { 0, 0, 0, 0, 0, 0, 0, 0 };       // unused trailing group (zeroed)
    __builtin_amdgcn_tensor_load_to_lds(g0, g1, g2, g3, g4, 0);
}

// ---------------- one LSTM timestep ----------------
// grid = 64 blocks (u-tiles of 16), block = 128 threads (4 waves).
// Wave w owns M rows [16w,16w+16) x u columns [u0,u0+16), all 4 gates.
// B (weights) staged in LDS by TDM, double-buffered per 256-wide K chunk.

__global__ __launch_bounds__(128)
void k_step(const _Float16* __restrict__ x16, const _Float16* __restrict__ Wp,
            const float* __restrict__ bias,
            const _Float16* __restrict__ h_in, _Float16* __restrict__ h_out,
            float* __restrict__ c32, float* __restrict__ h32, int t)
{
    extern __shared__ _Float16 smem[];   // 2 x 32KB chunk buffers (64KB dynamic)

    const int lane = threadIdx.x & 31;
    const int wave = threadIdx.x >> 5;     // 0..3 -> M-tile
    const int l16  = lane & 15;
    const int lh   = lane >> 4;            // lane-half: selects K sub-chunk
    const int u0   = blockIdx.x * 16;
    const int mrow = wave * 16 + l16;      // A-matrix row (batch index)

    v8f acc0 = {}, acc1 = {}, acc2 = {}, acc3 = {};   // gates i, f, c, o

    const _Float16* xrow = x16 + (size_t)mrow * (LSTM_T * LSTM_D) + (size_t)t * LSTM_D;
    const _Float16* hrow = h_in + (size_t)mrow * LSTM_U;

    const unsigned lds_base = (unsigned)__builtin_amdgcn_groupstaticsize();

    // Prologue: wave 0 kicks TDM for chunk 0 (4 gates).
    if (wave == 0) {
#pragma unroll
        for (int g = 0; g < 4; ++g)
            tdm_load_gate(lds_base + (unsigned)g * 8192u,
                          Wp + (size_t)(g * LSTM_U + u0) * LSTM_K);
    }

    for (int c = 0; c < NCHUNK; ++c) {
        if (wave == 0) {
            if (c + 1 < NCHUNK) {
                const unsigned boff = lds_base + (unsigned)((c + 1) & 1) * 32768u;
#pragma unroll
                for (int g = 0; g < 4; ++g)
                    tdm_load_gate(boff + (unsigned)g * 8192u,
                                  Wp + (size_t)(g * LSTM_U + u0) * LSTM_K + (c + 1) * KCHUNK);
                __builtin_amdgcn_s_wait_tensorcnt(4);  // in-order: chunk c landed
            } else {
                __builtin_amdgcn_s_wait_tensorcnt(0);
            }
        }
        __syncthreads();   // chunk c visible to all waves

        // B fragment base in LDS for this lane: row (g*16+l16), K-half lh
        const _Float16* lb = smem + (size_t)(c & 1) * CH_HALF + (size_t)l16 * KCHUNK + lh * 8;
        const int kbase = c * KCHUNK;

        v16h A = load_a(xrow, hrow, kbase, lh);
#pragma unroll
        for (int kk = 0; kk < KT_PER_CHUNK; ++kk) {
            v16h An;
            if (kk + 1 < KT_PER_CHUNK)
                An = load_a(xrow, hrow, kbase + (kk + 1) * 32, lh);
            const int ko = kk * 32;
            v16h B0 = load_b_lds(lb + 0 * 4096 + ko);
            v16h B1 = load_b_lds(lb + 1 * 4096 + ko);
            v16h B2 = load_b_lds(lb + 2 * 4096 + ko);
            v16h B3 = load_b_lds(lb + 3 * 4096 + ko);
            acc0 = WMMA_F16(A, B0, acc0);
            acc1 = WMMA_F16(A, B1, acc1);
            acc2 = WMMA_F16(A, B2, acc2);
            acc3 = WMMA_F16(A, B3, acc3);
            A = An;
        }
        __syncthreads();   // all waves done reading buf (c&1) before TDM reuses it
    }

    // Gate math. C/D layout: lane n = l16 (+u0), VGPR j holds m = 16*wave + j + 8*lh.
    const int u = u0 + l16;
    const float bi = bias[0 * LSTM_U + u];
    const float bf = bias[1 * LSTM_U + u];
    const float bc = bias[2 * LSTM_U + u];
    const float bo = bias[3 * LSTM_U + u];

#pragma unroll
    for (int j = 0; j < 8; ++j) {
        const int m = wave * 16 + lh * 8 + j;
        const size_t idx = (size_t)m * LSTM_U + u;
        const float gi = acc0[j] + bi;
        const float gf = acc1[j] + bf;
        const float gc = acc2[j] + bc;
        const float go = acc3[j] + bo;
        const float si = 1.0f / (1.0f + __expf(-gi));
        const float sf = 1.0f / (1.0f + __expf(-gf));
        const float so = 1.0f / (1.0f + __expf(-go));
        const float cn = sf * c32[idx] + si * tanhf(gc);
        const float hn = so * tanhf(cn);
        c32[idx]   = cn;
        h32[idx]   = hn;
        h_out[idx] = (_Float16)hn;
    }
}

// ---------------- finalize: out = [h, h, c] ----------------

__global__ void k_final(const float* __restrict__ h32, const float* __restrict__ c32,
                        float* __restrict__ out) {
    int i = blockIdx.x * 256 + threadIdx.x;  // B * 3U
    int b = i / (3 * LSTM_U);
    int j = i - b * (3 * LSTM_U);
    float v = (j < 2 * LSTM_U) ? h32[(size_t)b * LSTM_U + (j & (LSTM_U - 1))]
                               : c32[(size_t)b * LSTM_U + (j - 2 * LSTM_U)];
    out[i] = v;
}

// ---------------- launch ----------------

extern "C" void kernel_launch(void* const* d_in, const int* in_sizes, int n_in,
                              void* d_out, int out_size, void* d_ws, size_t ws_size,
                              hipStream_t stream) {
    const float* x  = (const float*)d_in[0];
    const float* Wx = (const float*)d_in[1];
    const float* Wh = (const float*)d_in[2];
    const float* b  = (const float*)d_in[3];

    char* ws = (char*)d_ws;
    // workspace layout (~45 MB total)
    _Float16* x16 = (_Float16*)(ws);                         // 64*512*512*2 = 33554432 B
    _Float16* Wp  = (_Float16*)(ws + 33554432);              // 4096*1536*2  = 12582912 B
    _Float16* hA  = (_Float16*)(ws + 46137344);              // 64*1024*2    =   131072 B
    _Float16* hB  = (_Float16*)(ws + 46268416);              //                 131072 B
    float*    c32 = (float*)   (ws + 46399488);              // 64*1024*4    =   262144 B
    float*    h32 = (float*)   (ws + 46661632);              //                 262144 B

    const int nx = LSTM_B * LSTM_T * LSTM_D;                 // 16777216
    k_cvt_x<<<nx / 256, 256, 0, stream>>>(x, x16, nx);
    k_pack_w<<<LSTM_4U, 256, 0, stream>>>(Wx, Wh, Wp);
    k_init<<<(LSTM_B * LSTM_U) / 256, 256, 0, stream>>>(hA, c32);

    for (int t = 0; t < LSTM_T; ++t) {
        const _Float16* hi = (t & 1) ? hB : hA;
        _Float16*       ho = (t & 1) ? hA : hB;
        k_step<<<LSTM_U / 16, 128, 65536, stream>>>(x16, Wp, b, hi, ho, c32, h32, t);
    }

    k_final<<<(LSTM_B * 3 * LSTM_U) / 256, 256, 0, stream>>>(h32, c32, (float*)d_out);
}
